// Attention_13348758356565
// MI455X (gfx1250) — matrix operand order repaired and checked by reading, not compiled, not run on hardware
//
#include <hip/hip_runtime.h>

// ---------------- problem constants ----------------
#define BB   16
#define HF   32
#define WF   32
#define NSEQ 1024          // HF*WF
#define DIM  1024
#define NH   16
#define HD   64            // DIM/NH
#define DIM3 3072          // 3*DIM
#define PI_F 3.14159265358979323846f

typedef __bf16 bh16;
typedef __attribute__((ext_vector_type(16))) __bf16 v16bf;
typedef __attribute__((ext_vector_type(8)))  __bf16 v8bf;
typedef __attribute__((ext_vector_type(8)))  float  v8f;
typedef __attribute__((ext_vector_type(4)))  unsigned int v4u;
typedef __attribute__((ext_vector_type(8)))  int v8i;
typedef __attribute__((ext_vector_type(4)))  int v4i;

#if defined(__HIP_DEVICE_COMPILE__) && \
    __has_builtin(__builtin_amdgcn_tensor_load_to_lds) && \
    __has_builtin(__builtin_amdgcn_s_wait_tensorcnt)
#define USE_TDM 1
#else
#define USE_TDM 0
#endif

static __device__ __forceinline__ v8f wmma_bf16(v16bf a, v16bf b, v8f c) {
    // D = A(16x32) * B(32x16) + C, f32 accum
    return __builtin_amdgcn_wmma_f32_16x16x32_bf16(
        /*neg_a=*/false, a, /*neg_b=*/false, b,
        /*c_mod=*/(short)0, c, /*reuse_a=*/false, /*reuse_b=*/false);
}

// A-fragment (16x32 bf16, MxK): lane = row (lane&15), half = lane>>4 selects
// K offsets {half*16 + 0..15} (contiguous, matching load_b_frag).
static __device__ __forceinline__ v16bf load_a_frag(const bh16* base, int ld,
                                                    int row, int k0, int half) {
    const bh16* p = base + (size_t)row * ld + k0 + half * 16;
    union { v16bf v; v8bf h[2]; } u;
    u.h[0] = *(const v8bf*)(p);
    u.h[1] = *(const v8bf*)(p + 8);
    return u.v;
}

// B-fragment (32x16 bf16, KxN) loaded from a row-major (N x K) matrix so the
// operand represents that matrix transposed: lane = column (lane&15),
// half selects K range {half*16 + 0..15} (contiguous 32B).
static __device__ __forceinline__ v16bf load_b_frag(const bh16* base, int ld,
                                                    int col, int k0, int half) {
    return *(const v16bf*)(base + (size_t)col * ld + k0 + half * 16);
}

#if USE_TDM
// Issue one 2D TDM tile load: tile (tile_d1 rows x tile_d0 elems, bf16) from a
// row-major tensor (row stride = stride0 elems) into LDS at lds_off, packed
// contiguously. D# bitfields per CDNA5 ISA §8.3/8.4 (data_size=1 -> 2 bytes).
// This toolchain exposes the 6-arg builtin: (g0, g1, g2, g3, g4, cpol).
static __device__ __forceinline__ void tdm_load_2d(const bh16* gaddr,
        unsigned lds_off, unsigned tensor_d0, unsigned tensor_d1,
        unsigned tile_d0, unsigned tile_d1, unsigned stride0) {
    unsigned long long ga = (unsigned long long)(uintptr_t)gaddr;
    v4u g0;
    g0.x = 1u;                                   // count=1 (valid descriptor)
    g0.y = lds_off;                              // lds_addr (bytes)
    g0.z = (unsigned)ga;                         // global_addr[31:0]
    g0.w = (unsigned)(ga >> 32) | (2u << 30);    // global_addr[56:32] | type=2
    v8i g1;
    g1[0] = (int)(1u << 16);                     // data_size=1 (2 bytes)
    g1[1] = (int)(tensor_d0 << 16);              // tensor_dim0[15:0] @ [63:48]
    g1[2] = (int)((tensor_d0 >> 16) | (tensor_d1 << 16)); // dim0 hi | dim1 lo
    g1[3] = (int)((tensor_d1 >> 16) | (tile_d0 << 16));   // dim1 hi | tile_dim0
    g1[4] = (int)(tile_d1 & 0xFFFFu);            // tile_dim1 (tile_dim2=0)
    g1[5] = (int)stride0;                        // tensor_dim0_stride[31:0]
    g1[6] = 0;                                   // stride0 hi | dim1_stride lo
    g1[7] = 0;
    v4i z4 = {};
    v8i z8 = {};
    __builtin_amdgcn_tensor_load_to_lds(g0, g1, z4, z4, z8, 0);
}
#endif

// ---------------- f32 -> bf16 conversion ----------------
__global__ void cvt_f32_bf16(const float* __restrict__ src,
                             bh16* __restrict__ dst, int n) {
    int i = blockIdx.x * blockDim.x + threadIdx.x;
    if (i < n) dst[i] = (bh16)src[i];
}

// ---------------- QKV projection GEMM ----------------
// C[16384,3072] = Xbf[16384,1024] * Wqkv[3072,1024]^T
// Each wave computes a 64x32 tile (4x2 WMMA accumulators).
// Epilogue scatters into Q/K (B,NH,N,HD) and V^T (B,NH,HD,N) via selects.
__global__ __launch_bounds__(256) void gemm_qkv_kernel(
        const bh16* __restrict__ X, const bh16* __restrict__ W,
        bh16* __restrict__ Q, bh16* __restrict__ K, bh16* __restrict__ Vt) {
    const int lane = threadIdx.x & 31;
    const int l15  = lane & 15;
    const int half = lane >> 4;
    const int wave = threadIdx.x >> 5;
    const int NT   = DIM3 / 32;                 // 96 col tiles
    int tile = blockIdx.x * 8 + wave;           // 3072 blocks * 8 = 24576 tiles
    int tm = tile / NT, tn = tile % NT;
    int r0 = tm * 64, c0 = tn * 32;

    v8f acc[4][2] = {};
    for (int k0 = 0; k0 < DIM; k0 += 32) {
        v16bf a[4], b[2];
#pragma unroll
        for (int mi = 0; mi < 4; ++mi)
            a[mi] = load_a_frag(X, DIM, r0 + mi * 16 + l15, k0, half);
#pragma unroll
        for (int ni = 0; ni < 2; ++ni)
            b[ni] = load_b_frag(W, DIM, c0 + ni * 16 + l15, k0, half);
#pragma unroll
        for (int mi = 0; mi < 4; ++mi)
#pragma unroll
            for (int ni = 0; ni < 2; ++ni)
                acc[mi][ni] = wmma_bf16(a[mi], b[ni], acc[mi][ni]);
    }

    const int bidx = r0 >> 10;                  // batch (r0 multiple of 64)
#pragma unroll
    for (int ni = 0; ni < 2; ++ni) {
        int col = c0 + ni * 16 + l15;
        int qkv = col >> 10;
        int rem = col & (DIM - 1);
        int h   = rem >> 6;
        int d   = rem & (HD - 1);
        size_t bh = (size_t)(bidx * NH + h);
        // branch-free scatter: select base pointer + linear index formula
        bh16* base = (qkv == 0) ? Q : ((qkv == 1) ? K : Vt);
        bool isV = (qkv == 2);
        size_t idx0 = isV ? ((bh * HD + d) * NSEQ)       // + n*1
                          : (bh * NSEQ * HD + d);        // + n*HD
        size_t step = isV ? (size_t)1 : (size_t)HD;
#pragma unroll
        for (int mi = 0; mi < 4; ++mi) {
#pragma unroll
            for (int i = 0; i < 8; ++i) {
                int row = r0 + mi * 16 + half * 8 + i;
                int n   = row & (NSEQ - 1);
                base[idx0 + (size_t)n * step] = (bh16)acc[mi][ni][i];
            }
        }
    }
}

// ---------------- axial RoPE (in place on Q and K) ----------------
// rotary dim = HD/2 = 32 -> 16 pairs; pairs 0..7 use posH(y), 8..15 use posW(x)
__global__ void rope_kernel(bh16* __restrict__ Q, bh16* __restrict__ K) {
    int t = blockIdx.x * blockDim.x + threadIdx.x;   // B*NH*N*16 threads
    int p  = t & 15;
    int n  = (t >> 4) & (NSEQ - 1);
    int bh = t >> 14;
    if (bh >= BB * NH) return;
    int y = n >> 5, x = n & 31;
    int j = p & 7;
    float base = (1.0f + (float)j * (511.0f / 7.0f)) * PI_F; // linspace(1,512,8)*pi
    float pos  = (p < 8) ? (-1.0f + (2.0f / 31.0f) * (float)y)
                         : (-1.0f + (2.0f / 31.0f) * (float)x);
    float f = pos * base;
    float c = __cosf(f), s = __sinf(f);
    size_t idx = ((size_t)bh * NSEQ + n) * HD + 2 * p;
    float q0 = (float)Q[idx], q1 = (float)Q[idx + 1];
    Q[idx]     = (bh16)(q0 * c - q1 * s);
    Q[idx + 1] = (bh16)(q1 * c + q0 * s);
    float k0 = (float)K[idx], k1 = (float)K[idx + 1];
    K[idx]     = (bh16)(k0 * c - k1 * s);
    K[idx + 1] = (bh16)(k1 * c + k0 * s);
}

// ---------------- flash attention ----------------
// One block (8 waves) per (b,h, 128-row q chunk); each wave owns 16 q rows.
// K/V tiles for the block are staged into LDS by the Tensor Data Mover
// (double-buffered, s_wait_tensorcnt), shared by all 8 waves.
// Online softmax; P re-laid out C-frag -> A-frag through 1KB LDS per wave.
__global__ __launch_bounds__(256) void attn_kernel(
        const bh16* __restrict__ Q, const bh16* __restrict__ K,
        const bh16* __restrict__ Vt, bh16* __restrict__ O) {
    __shared__ bh16 Plds[8 * 16 * 32];
#if USE_TDM
    __shared__ bh16 Klds[2][32 * 64];   // 32 keys x 64 d, packed
    __shared__ bh16 Vlds[2][64 * 32];   // 64 d x 32 keys, packed
#endif
    const int lane = threadIdx.x & 31;
    const int l15  = lane & 15;
    const int half = lane >> 4;
    const int wave = threadIdx.x >> 5;
    const int bh   = blockIdx.x >> 3;
    const int qc   = blockIdx.x & 7;
    const int m0   = qc * 128 + wave * 16;

    const bh16* Qp = Q  + (size_t)bh * NSEQ * HD;
    const bh16* Kp = K  + (size_t)bh * NSEQ * HD;
    const bh16* Vp = Vt + (size_t)bh * HD * NSEQ;
    bh16* Pw = Plds + wave * 512;

    v16bf qa0 = load_a_frag(Qp, HD, m0 + l15, 0,  half);
    v16bf qa1 = load_a_frag(Qp, HD, m0 + l15, 32, half);

    v8f o[4] = {};
    float mrow[8], lrow[8];
#pragma unroll
    for (int i = 0; i < 8; ++i) { mrow[i] = -1e30f; lrow[i] = 0.0f; }

#if USE_TDM
    // scalar (SGPR) wave id so TDM issue is guarded by a *scalar* branch:
    // TDM ignores EXEC, so exec-masking would not suppress duplicate issues.
    const int wave_u = __builtin_amdgcn_readfirstlane(wave);
    unsigned koff[2] = { (unsigned)(uintptr_t)&Klds[0][0],
                         (unsigned)(uintptr_t)&Klds[1][0] };
    unsigned voff[2] = { (unsigned)(uintptr_t)&Vlds[0][0],
                         (unsigned)(uintptr_t)&Vlds[1][0] };
    if (wave_u == 0) {
        // prime buffer 0 with kb = 0
        tdm_load_2d(Kp,      koff[0], HD, NSEQ, HD, 32, HD);    // 32 key rows
        tdm_load_2d(Vp,      voff[0], NSEQ, HD, 32, HD, NSEQ);  // 64 d rows
    }
    int cur = 0;
#endif

    const float scale = 0.125f;                       // 1/sqrt(64)
    for (int kb = 0; kb < NSEQ; kb += 32) {
#if USE_TDM
        if (wave_u == 0) {
            if (kb + 32 < NSEQ) {
                int nxt = cur ^ 1;
                tdm_load_2d(Kp + (size_t)(kb + 32) * HD, koff[nxt],
                            HD, NSEQ, HD, 32, HD);
                tdm_load_2d(Vp + (kb + 32),              voff[nxt],
                            NSEQ, HD, 32, HD, NSEQ);
                __builtin_amdgcn_s_wait_tensorcnt(2);  // current buffer ready
            } else {
                __builtin_amdgcn_s_wait_tensorcnt(0);
            }
        }
        __syncthreads();                               // tiles visible to all
        const bh16* Kt = &Klds[cur][0];                // 32 x 64, ld=64
        const bh16* Vtile = &Vlds[cur][0];             // 64 x 32, ld=32
        v16bf kf00 = load_b_frag(Kt, HD, l15,      0,  half);
        v16bf kf01 = load_b_frag(Kt, HD, l15,      32, half);
        v16bf kf10 = load_b_frag(Kt, HD, 16 + l15, 0,  half);
        v16bf kf11 = load_b_frag(Kt, HD, 16 + l15, 32, half);
#else
        v16bf kf00 = load_b_frag(Kp, HD, kb + l15,      0,  half);
        v16bf kf01 = load_b_frag(Kp, HD, kb + l15,      32, half);
        v16bf kf10 = load_b_frag(Kp, HD, kb + 16 + l15, 0,  half);
        v16bf kf11 = load_b_frag(Kp, HD, kb + 16 + l15, 32, half);
#endif
        v8f z = {};
        v8f s1 = wmma_bf16(qa0, kf00, wmma_bf16(qa1, kf01, z));
        v8f s2 = wmma_bf16(qa0, kf10, wmma_bf16(qa1, kf11, z));

#pragma unroll
        for (int i = 0; i < 8; ++i) {
            float a  = s1[i] * scale;
            float b2 = s2[i] * scale;
            float tm = fmaxf(a, b2);
            tm = fmaxf(tm, __shfl_xor(tm, 1, 32));
            tm = fmaxf(tm, __shfl_xor(tm, 2, 32));
            tm = fmaxf(tm, __shfl_xor(tm, 4, 32));
            tm = fmaxf(tm, __shfl_xor(tm, 8, 32));
            float nm = fmaxf(mrow[i], tm);
            float e1 = __expf(a - nm);
            float e2 = __expf(b2 - nm);
            float rs = e1 + e2;
            rs += __shfl_xor(rs, 1, 32);
            rs += __shfl_xor(rs, 2, 32);
            rs += __shfl_xor(rs, 4, 32);
            rs += __shfl_xor(rs, 8, 32);
            float corr = __expf(mrow[i] - nm);
            lrow[i] = lrow[i] * corr + rs;
            mrow[i] = nm;
#pragma unroll
            for (int t = 0; t < 4; ++t) o[t][i] *= corr;
            int row = half * 8 + i;
            Pw[row * 32 + l15]      = (bh16)e1;
            Pw[row * 32 + 16 + l15] = (bh16)e2;
        }
        __syncthreads();
        v16bf pa = load_a_frag(Pw, 32, l15, 0, half);
#if USE_TDM
        v16bf v0 = load_b_frag(Vtile, 32, 0 * 16 + l15, 0, half);
        v16bf v1 = load_b_frag(Vtile, 32, 1 * 16 + l15, 0, half);
        v16bf v2 = load_b_frag(Vtile, 32, 2 * 16 + l15, 0, half);
        v16bf v3 = load_b_frag(Vtile, 32, 3 * 16 + l15, 0, half);
#else
        v16bf v0 = load_b_frag(Vp, NSEQ, 0 * 16 + l15, kb, half);
        v16bf v1 = load_b_frag(Vp, NSEQ, 1 * 16 + l15, kb, half);
        v16bf v2 = load_b_frag(Vp, NSEQ, 2 * 16 + l15, kb, half);
        v16bf v3 = load_b_frag(Vp, NSEQ, 3 * 16 + l15, kb, half);
#endif
        o[0] = wmma_bf16(pa, v0, o[0]);
        o[1] = wmma_bf16(pa, v1, o[1]);
        o[2] = wmma_bf16(pa, v2, o[2]);
        o[3] = wmma_bf16(pa, v3, o[3]);
        __syncthreads();
#if USE_TDM
        cur ^= 1;
#endif
    }

    const int b = bh >> 4, h = bh & 15;
#pragma unroll
    for (int t = 0; t < 4; ++t) {
#pragma unroll
        for (int i = 0; i < 8; ++i) {
            int nq = m0 + half * 8 + i;
            size_t row = (size_t)b * NSEQ + nq;
            int col = h * HD + t * 16 + l15;
            O[row * DIM + col] = (bh16)(o[t][i] / lrow[i]);
        }
    }
}

// ---------------- output projection GEMM + bias ----------------
// out[16384,1024] = Obf[16384,1024] * Wproj[1024,1024]^T + b, f32 out
__global__ __launch_bounds__(256) void gemm_proj_kernel(
        const bh16* __restrict__ A, const bh16* __restrict__ W,
        const float* __restrict__ bias, float* __restrict__ out) {
    const int lane = threadIdx.x & 31;
    const int l15  = lane & 15;
    const int half = lane >> 4;
    const int wave = threadIdx.x >> 5;
    const int NT   = DIM / 32;                  // 32 col tiles
    int tile = blockIdx.x * 8 + wave;           // 1024 blocks * 8 = 8192 tiles
    int tm = tile / NT, tn = tile % NT;
    int r0 = tm * 64, c0 = tn * 32;

    v8f acc[4][2] = {};
    for (int k0 = 0; k0 < DIM; k0 += 32) {
        v16bf a[4], b[2];
#pragma unroll
        for (int mi = 0; mi < 4; ++mi)
            a[mi] = load_a_frag(A, DIM, r0 + mi * 16 + l15, k0, half);
#pragma unroll
        for (int ni = 0; ni < 2; ++ni)
            b[ni] = load_b_frag(W, DIM, c0 + ni * 16 + l15, k0, half);
#pragma unroll
        for (int mi = 0; mi < 4; ++mi)
#pragma unroll
            for (int ni = 0; ni < 2; ++ni)
                acc[mi][ni] = wmma_bf16(a[mi], b[ni], acc[mi][ni]);
    }
#pragma unroll
    for (int ni = 0; ni < 2; ++ni) {
        int col = c0 + ni * 16 + l15;
        float bb = bias[col];
#pragma unroll
        for (int mi = 0; mi < 4; ++mi) {
#pragma unroll
            for (int i = 0; i < 8; ++i) {
                int row = r0 + mi * 16 + half * 8 + i;
                out[(size_t)row * DIM + col] = acc[mi][ni][i] + bb;
            }
        }
    }
}

// ---------------- host launcher ----------------
extern "C" void kernel_launch(void* const* d_in, const int* in_sizes, int n_in,
                              void* d_out, int out_size, void* d_ws, size_t ws_size,
                              hipStream_t stream) {
    (void)in_sizes; (void)n_in; (void)out_size; (void)ws_size;
    const float* x      = (const float*)d_in[0];
    const float* w_qkv  = (const float*)d_in[1];
    const float* w_proj = (const float*)d_in[2];
    const float* b_proj = (const float*)d_in[3];
    float* out = (float*)d_out;

    char* ws = (char*)d_ws;
    size_t off = 0;
    bh16* Xbf     = (bh16*)(ws + off); off += (size_t)BB * NSEQ * DIM * 2;   // 32 MB
    bh16* Wqkvbf  = (bh16*)(ws + off); off += (size_t)DIM3 * DIM * 2;        //  6 MB
    bh16* Wprojbf = (bh16*)(ws + off); off += (size_t)DIM * DIM * 2;         //  2 MB
    bh16* Qbf     = (bh16*)(ws + off); off += (size_t)BB * NH * NSEQ * HD * 2; // 32 MB
    bh16* Kbf     = (bh16*)(ws + off); off += (size_t)BB * NH * NSEQ * HD * 2; // 32 MB
    bh16* Vtbf    = (bh16*)(ws + off); off += (size_t)BB * NH * HD * NSEQ * 2; // 32 MB
    bh16* Obf     = (bh16*)(ws + off); off += (size_t)BB * NSEQ * DIM * 2;   // 32 MB

    const int nx = BB * NSEQ * DIM;       // 16777216
    const int nq = DIM3 * DIM;            // 3145728
    const int np = DIM * DIM;             // 1048576
    cvt_f32_bf16<<<(nx + 255) / 256, 256, 0, stream>>>(x, Xbf, nx);
    cvt_f32_bf16<<<(nq + 255) / 256, 256, 0, stream>>>(w_qkv, Wqkvbf, nq);
    cvt_f32_bf16<<<(np + 255) / 256, 256, 0, stream>>>(w_proj, Wprojbf, np);

    // 16384/64 * 3072/32 = 24576 wave-tiles / 8 waves per block
    gemm_qkv_kernel<<<3072, 256, 0, stream>>>(Xbf, Wqkvbf, Qbf, Kbf, Vtbf);

    // B*NH*N*16 pair-threads
    rope_kernel<<<(BB * NH * NSEQ * 16) / 256, 256, 0, stream>>>(Qbf, Kbf);

    // B*NH * (N/128) blocks
    attn_kernel<<<BB * NH * (NSEQ / 128), 256, 0, stream>>>(Qbf, Kbf, Vtbf, Obf);

    // 16384/64 * 1024/32 = 8192 wave-tiles / 8 waves per block
    gemm_proj_kernel<<<1024, 256, 0, stream>>>(Obf, Wprojbf, b_proj, out);
}
